// PointerAttention_11871289606763
// MI455X (gfx1250) — matrix operand order repaired and checked by reading, not compiled, not run on hardware
//
#include <hip/hip_runtime.h>
#include <math.h>

#define B_ 8
#define S_ 8192
#define D_ 256
#define BIGNUM 1000000.0f

typedef __attribute__((ext_vector_type(16))) __bf16       v16bf;
typedef __attribute__((ext_vector_type(8)))  float        v8f;
typedef __attribute__((ext_vector_type(4)))  unsigned int u32x4;
typedef __attribute__((ext_vector_type(4)))  float        f32x4;

union FragBF16 { u32x4 q[2]; v16bf v; };

__device__ __forceinline__ unsigned int f32_to_bf16_bits(float f) {
    unsigned int u = __float_as_uint(f);
    unsigned int lsb = (u >> 16) & 1u;
    u += 0x7fffu + lsb;                 // round-to-nearest-even
    return u >> 16;
}

// Branch-free tanh: 1 - 2/(exp(2x)+1). exp overflow/underflow saturates to +/-1.
// v_mul + v_exp(trans) + v_add + v_rcp(trans) + v_fma -- no divergence.
__device__ __forceinline__ float fast_tanh(float x) {
    float e = __expf(2.0f * x);
    float r = __builtin_amdgcn_rcpf(e + 1.0f);
    return __builtin_fmaf(-2.0f, r, 1.0f);
}

// ---------------------------------------------------------------------------
// Kernel 1: dec_proj[b,n] = sum_k dec[b,k]*W1[k,n] + W1_b[n] + W2_b[n]
// ---------------------------------------------------------------------------
__global__ __launch_bounds__(D_)
void decproj_kernel(const float* __restrict__ dec, const float* __restrict__ W1,
                    const float* __restrict__ W1b, const float* __restrict__ W2b,
                    float* __restrict__ dp) {
    const int b = blockIdx.x, n = threadIdx.x;
    float s = W1b[n] + W2b[n];
    const float* drow = dec + b * D_;
    for (int k = 0; k < D_; ++k) s += drow[k] * W1[k * D_ + n];
    dp[b * D_ + n] = s;
}

// ---------------------------------------------------------------------------
// Kernel 2: Wt[n,k] = bf16(W2[k,n])   (transposed so B-fragments are K-contig)
// ---------------------------------------------------------------------------
__global__ __launch_bounds__(D_)
void w2t_kernel(const float* __restrict__ W2, unsigned short* __restrict__ Wt) {
    const int n = blockIdx.x, k = threadIdx.x;
    Wt[n * D_ + k] = (unsigned short)f32_to_bf16_bits(W2[k * D_ + n]);
}

// ---------------------------------------------------------------------------
// Kernel 3 (fused): enc_proj GEMM via WMMA bf16 + tanh + dot(V_k) + mask
// Block: 256 threads (8 waves), 128 rows of S per block, grid (S/128, B).
// LDS: W2^T only (128 KB) + dp/vk (2 KB) -> 2 workgroups per WGP.
// A fragments are loaded straight from global (f32) and converted in regs.
// ---------------------------------------------------------------------------
__global__ __launch_bounds__(256)
void fused_score_kernel(const float* __restrict__ enc_outputs,
                        const float* __restrict__ mask,
                        const float* __restrict__ dp_g,
                        const unsigned short* __restrict__ Wt_g,
                        const float* __restrict__ Vk,
                        const float* __restrict__ Vb,
                        float* __restrict__ logits) {
    extern __shared__ char smem[];
    unsigned short* Wt_s = (unsigned short*)smem;               // 128 KB
    float* dp_s = (float*)(smem + D_ * D_ * 2);                 //   1 KB
    float* vk_s = dp_s + D_;                                    //   1 KB

    const int tid = threadIdx.x;
    const int b   = blockIdx.y;
    const int s0  = blockIdx.x * 128;

    // Stage W2^T (bf16, 256x256) into LDS: 8192 x 16B chunks.
    {
        const u32x4* src = (const u32x4*)Wt_g;
        u32x4*       dst = (u32x4*)Wt_s;
        for (int i = tid; i < (D_ * D_) / 8; i += 256) dst[i] = src[i];
    }
    if (tid < D_) {
        dp_s[tid] = dp_g[b * D_ + tid];
        vk_s[tid] = Vk[tid];
    }

    const int wv   = tid >> 5;     // wave id: owns rows [16*wv, 16*wv+16)
    const int lane = tid & 31;
    const int mn   = lane & 15;    // M for A-frag, N for B/C-frag
    const int hf   = lane >> 4;    // lane half

    // Load this wave's A fragments directly from global, f32 -> bf16 in regs.
    // 16-bit A 16x32 layout: j<8 -> K = 8*hf + j ; j>=8 -> K = 16 + 8*hf + (j-8)
    const float* arow = enc_outputs + ((size_t)b * S_ + s0 + wv * 16 + mn) * D_;
    FragBF16 afrag[8];
    #pragma unroll
    for (int kt = 0; kt < 8; ++kt) {
        const float* p0 = arow + kt * 32 + hf * 8;
        f32x4 a0 = *(const f32x4*)p0;            // j = 0..3
        f32x4 a1 = *(const f32x4*)(p0 + 4);      // j = 4..7
        f32x4 a2 = *(const f32x4*)(p0 + 16);     // j = 8..11
        f32x4 a3 = *(const f32x4*)(p0 + 20);     // j = 12..15
        u32x4 q0, q1;
        q0.x = f32_to_bf16_bits(a0.x) | (f32_to_bf16_bits(a0.y) << 16);
        q0.y = f32_to_bf16_bits(a0.z) | (f32_to_bf16_bits(a0.w) << 16);
        q0.z = f32_to_bf16_bits(a1.x) | (f32_to_bf16_bits(a1.y) << 16);
        q0.w = f32_to_bf16_bits(a1.z) | (f32_to_bf16_bits(a1.w) << 16);
        q1.x = f32_to_bf16_bits(a2.x) | (f32_to_bf16_bits(a2.y) << 16);
        q1.y = f32_to_bf16_bits(a2.z) | (f32_to_bf16_bits(a2.w) << 16);
        q1.z = f32_to_bf16_bits(a3.x) | (f32_to_bf16_bits(a3.y) << 16);
        q1.w = f32_to_bf16_bits(a3.z) | (f32_to_bf16_bits(a3.w) << 16);
        afrag[kt].q[0] = q0;
        afrag[kt].q[1] = q1;
    }

    __syncthreads();   // Wt_s / dp_s / vk_s ready

    float partial[8];
    #pragma unroll
    for (int r = 0; r < 8; ++r) partial[r] = 0.0f;

    for (int nt = 0; nt < 16; ++nt) {
        v8f acc = {};
        // B 32x16 layout: N = lane%16, j<8 -> K = 8*hf + j ; j>=8 -> K = 16 + 8*hf + (j-8)
        const unsigned short* bp0 = Wt_s + (nt * 16 + mn) * D_ + hf * 8;
        #pragma unroll
        for (int kt = 0; kt < 8; ++kt) {
            FragBF16 bfrag;
            const unsigned short* bp = bp0 + kt * 32;
            bfrag.q[0] = *(const u32x4*)bp;
            bfrag.q[1] = *(const u32x4*)(bp + 16);
            acc = __builtin_amdgcn_wmma_f32_16x16x32_bf16(
                false, afrag[kt].v, false, bfrag.v, (short)0, acc, false, false);
        }
        // Fused epilogue: C layout lane%16 = N, VGPR r -> row M = r + 8*hf
        const int   n   = nt * 16 + mn;
        const float dpn = dp_s[n];
        const float vkn = vk_s[n];
        #pragma unroll
        for (int r = 0; r < 8; ++r)
            partial[r] += fast_tanh(acc[r] + dpn) * vkn;
    }

    // Reduce over the 16 lanes sharing each row (xor 1,2,4,8 stays in half).
    #pragma unroll
    for (int r = 0; r < 8; ++r) {
        float v = partial[r];
        v += __shfl_xor(v, 1, 32);
        v += __shfl_xor(v, 2, 32);
        v += __shfl_xor(v, 4, 32);
        v += __shfl_xor(v, 8, 32);
        partial[r] = v;
    }
    const float vb = Vb[0];
    if (mn == 0) {                               // lanes 0 (rows 0..7) and 16 (rows 8..15)
        const int srow = s0 + wv * 16 + hf * 8;
        #pragma unroll
        for (int r = 0; r < 8; ++r) {
            const int s = srow + r;
            logits[(size_t)b * S_ + s] =
                partial[r] + vb - mask[(size_t)b * S_ + s] * BIGNUM;
        }
    }
}

// ---------------------------------------------------------------------------
// Kernel 4: per-batch softmax over S + argmax for last batch row
// ---------------------------------------------------------------------------
__global__ __launch_bounds__(1024)
void softmax_kernel(const float* __restrict__ logits, float* __restrict__ probs,
                    float* __restrict__ idx_out, int* __restrict__ idx_ws) {
    __shared__ float sred[32];
    __shared__ int   sarg[32];
    const int b = blockIdx.x, tid = threadIdx.x;
    const int wv = tid >> 5, ln = tid & 31;
    const float* row = logits + (size_t)b * S_;

    float lmax = -3.4e38f; int larg = 0;
    for (int s = tid; s < S_; s += 1024) {
        float v = row[s];
        if (v > lmax) { lmax = v; larg = s; }
    }
    #pragma unroll
    for (int m = 16; m >= 1; m >>= 1) {
        float ov = __shfl_xor(lmax, m, 32);
        int   oi = __shfl_xor(larg, m, 32);
        if (ov > lmax || (ov == lmax && oi < larg)) { lmax = ov; larg = oi; }
    }
    if (ln == 0) { sred[wv] = lmax; sarg[wv] = larg; }
    __syncthreads();
    if (wv == 0) {
        float v = sred[ln]; int i = sarg[ln];
        #pragma unroll
        for (int m = 16; m >= 1; m >>= 1) {
            float ov = __shfl_xor(v, m, 32);
            int   oi = __shfl_xor(i, m, 32);
            if (ov > v || (ov == v && oi < i)) { v = ov; i = oi; }
        }
        if (ln == 0) { sred[0] = v; sarg[0] = i; }
    }
    __syncthreads();
    const float bmax = sred[0];
    const int   barg = sarg[0];
    __syncthreads();

    float lsum = 0.f;
    for (int s = tid; s < S_; s += 1024) lsum += __expf(row[s] - bmax);
    #pragma unroll
    for (int m = 16; m >= 1; m >>= 1) lsum += __shfl_xor(lsum, m, 32);
    if (ln == 0) sred[wv] = lsum;
    __syncthreads();
    if (wv == 0) {
        float v = sred[ln];
        #pragma unroll
        for (int m = 16; m >= 1; m >>= 1) v += __shfl_xor(v, m, 32);
        if (ln == 0) sred[0] = v;
    }
    __syncthreads();
    const float inv = __builtin_amdgcn_rcpf(sred[0]);
    for (int s = tid; s < S_; s += 1024)
        probs[(size_t)b * S_ + s] = __expf(row[s] - bmax) * inv;

    if (b == B_ - 1 && tid == 0) { *idx_ws = barg; *idx_out = (float)barg; }
}

// ---------------------------------------------------------------------------
// Kernel 5: dec_out[b,:] = enc_input[b, idx, :]
// ---------------------------------------------------------------------------
__global__ __launch_bounds__(D_)
void gather_kernel(const float* __restrict__ enc_input,
                   const int* __restrict__ idx, float* __restrict__ outp) {
    const int b = blockIdx.x, d = threadIdx.x;
    const int p = *idx;
    outp[b * D_ + d] = enc_input[((size_t)b * S_ + p) * D_ + d];
}

extern "C" void kernel_launch(void* const* d_in, const int* in_sizes, int n_in,
                              void* d_out, int out_size, void* d_ws, size_t ws_size,
                              hipStream_t stream) {
    const float* dec_output  = (const float*)d_in[0];
    const float* enc_input   = (const float*)d_in[1];
    const float* enc_outputs = (const float*)d_in[2];
    const float* mask        = (const float*)d_in[3];
    const float* W1_k = (const float*)d_in[4];
    const float* W1_b = (const float*)d_in[5];
    const float* W2_k = (const float*)d_in[6];
    const float* W2_b = (const float*)d_in[7];
    const float* V_k  = (const float*)d_in[8];
    const float* V_b  = (const float*)d_in[9];

    float* out    = (float*)d_out;
    float* logits = out;                          // [B,S]
    float* probs  = out + (size_t)B_ * S_;        // [B,S]
    float* idxf   = out + 2 * (size_t)B_ * S_;    // scalar
    float* decout = idxf + 1;                     // [B,D]

    char* ws = (char*)d_ws;
    float*          dp_ws  = (float*)ws;                                  // B*D f32
    unsigned short* Wt_ws  = (unsigned short*)(ws + B_ * D_ * 4);         // D*D bf16
    int*            idx_ws = (int*)(ws + B_ * D_ * 4 + D_ * D_ * 2);

    decproj_kernel<<<B_, D_, 0, stream>>>(dec_output, W1_k, W1_b, W2_b, dp_ws);
    w2t_kernel<<<D_, D_, 0, stream>>>(W2_k, Wt_ws);

    const size_t lds_bytes = (size_t)D_ * D_ * 2 + 2 * D_ * 4;  // 133120 B
    fused_score_kernel<<<dim3(S_ / 128, B_), 256, lds_bytes, stream>>>(
        enc_outputs, mask, dp_ws, Wt_ws, V_k, V_b, logits);

    softmax_kernel<<<B_, 1024, 0, stream>>>(logits, probs, idxf, idx_ws);
    gather_kernel<<<B_, D_, 0, stream>>>(enc_input, idx_ws, decout);
}
